// GPPO_25692494364720
// MI455X (gfx1250) — matrix operand order repaired and checked by reading, not compiled, not run on hardware
//
#include <hip/hip_runtime.h>
#include <hip/hip_bf16.h>
#include <math.h>

typedef __attribute__((ext_vector_type(16))) _Float16 v16h;
typedef __attribute__((ext_vector_type(8)))  _Float16 v8h;
typedef __attribute__((ext_vector_type(8)))  float    v8f;

#define LDS_FENCE() __asm__ volatile("s_wait_dscnt 0" ::: "memory")

#define WMMA_F16(A, B, C) \
    __builtin_amdgcn_wmma_f32_16x16x32_f16(false, (A), false, (B), (short)0, (C), false, false)

// branch-free tanh: 1 - 2/(e^{2x}+1); saturates correctly at +-inf via v_exp/v_rcp
static __device__ __forceinline__ float fast_tanh(float x) {
    float e = __expf(2.0f * x);
    return 1.0f - 2.0f * __builtin_amdgcn_rcpf(e + 1.0f);
}

// ---- WMMA fragment loaders (CDNA5 16-bit A 16x32 / B 32x16 layouts) ----
// A element i of v16h: K = k0 + 8*hs + (i<8 ? i : 8+i)  [two runs of 8 consecutive K]

// A fragment from global f32, row-major [rows, ld]; KMAX compile-time -> guards fold.
template <int KMAX>
static __device__ __forceinline__ v16h load_a_g(const float* __restrict__ src, int ld,
                                                int row_base, int k0) {
    int lane = threadIdx.x & 31;
    int m = lane & 15, hs = lane >> 4;
    const float* rp = src + (size_t)(row_base + m) * (size_t)ld;
    float4 r0a = {0.f, 0.f, 0.f, 0.f}, r0b = r0a, r1a = r0a, r1b = r0a;
    if (k0 + 16 <= KMAX) {  // run0 fully valid for both hs (K multiples of 8 only)
        r0a = *(const float4*)(rp + k0 + 8 * hs);
        r0b = *(const float4*)(rp + k0 + 8 * hs + 4);
    }
    if (k0 + 32 <= KMAX) {  // run1 fully valid
        r1a = *(const float4*)(rp + k0 + 16 + 8 * hs);
        r1b = *(const float4*)(rp + k0 + 16 + 8 * hs + 4);
    }
    v16h a;
    a[0] = (_Float16)r0a.x;  a[1] = (_Float16)r0a.y;  a[2] = (_Float16)r0a.z;  a[3] = (_Float16)r0a.w;
    a[4] = (_Float16)r0b.x;  a[5] = (_Float16)r0b.y;  a[6] = (_Float16)r0b.z;  a[7] = (_Float16)r0b.w;
    a[8] = (_Float16)r1a.x;  a[9] = (_Float16)r1a.y;  a[10] = (_Float16)r1a.z; a[11] = (_Float16)r1a.w;
    a[12] = (_Float16)r1b.x; a[13] = (_Float16)r1b.y; a[14] = (_Float16)r1b.z; a[15] = (_Float16)r1b.w;
    return a;
}

// A fragment from LDS f16, row-major [16][80] (rows 160B -> 16B-aligned runs)
static __device__ __forceinline__ v16h load_a_s(const _Float16* sp, int k0) {
    int lane = threadIdx.x & 31;
    int m = lane & 15, hs = lane >> 4;
    const _Float16* p = sp + m * 80 + k0 + 8 * hs;
    v8h lo = *(const v8h*)(p);
    v8h hi = *(const v8h*)(p + 16);
    return __builtin_shufflevector(lo, hi, 0, 1, 2, 3, 4, 5, 6, 7, 8, 9, 10, 11, 12, 13, 14, 15);
}

// B fragment from pre-transposed f16 weights wt[n][k] (ldk = padded K, multiple of 32)
static __device__ __forceinline__ v16h load_b_t(const _Float16* __restrict__ wt, int ldk,
                                                int k0, int n0) {
    int lane = threadIdx.x & 31;
    int n = lane & 15, kh = lane >> 4;
    const _Float16* p = wt + (size_t)(n0 + n) * ldk + k0 + 8 * kh;
    v8h lo = *(const v8h*)(p);
    v8h hi = *(const v8h*)(p + 16);
    return __builtin_shufflevector(lo, hi, 0, 1, 2, 3, 4, 5, 6, 7, 8, 9, 10, 11, 12, 13, 14, 15);
}

// ---- elementwise / utility kernels ----

__global__ void k_zero(float* __restrict__ p, int n) {
    int i = blockIdx.x * blockDim.x + threadIdx.x;
    if (i < n) p[i] = 0.0f;
}

__global__ void k_copy(const float* __restrict__ s, float* __restrict__ d, long n) {
    long i = (long)blockIdx.x * blockDim.x + threadIdx.x;
    long stride = (long)gridDim.x * blockDim.x;
    for (; i < n; i += stride) d[i] = s[i];
}

// transpose + f16-convert weights: wt[n*Kpad + k] = (f16) w[k*ncols + n], zero pad k>=K
__global__ void k_wprep(const float* __restrict__ w, _Float16* __restrict__ wt,
                        int K, int Kpad, int ncols) {
    int i = blockIdx.x * blockDim.x + threadIdx.x;
    if (i >= ncols * Kpad) return;
    int n = i / Kpad, k = i % Kpad;
    wt[i] = (_Float16)((k < K) ? w[k * ncols + n] : 0.0f);
}

// scatter-add over edges: agg[dst] += hin[src]; one thread per (edge, 4-col group)
__global__ void k_edge(const float* __restrict__ hin, const int* __restrict__ src,
                       const int* __restrict__ dst, float* __restrict__ agg,
                       int K, int qshift, int E_) {
    long t = (long)blockIdx.x * blockDim.x + threadIdx.x;
    int e = (int)(t >> qshift);
    int q = (int)(t & ((1 << qshift) - 1));
    if (e >= E_) return;
    int s = src[e], d = dst[e];
    const float4 v = *(const float4*)(hin + (size_t)s * K + q * 4);
    float* ap = agg + (size_t)d * K + q * 4;
    atomicAdd(ap + 0, v.x);
    atomicAdd(ap + 1, v.y);
    atomicAdd(ap + 2, v.z);
    atomicAdd(ap + 3, v.w);
}

// ---- GIN MLP: Hout = relu(Ain @ w1 + b1) @ w2 + b2 ; accumulates BN stats ----
template <int KIN>
__global__ void __launch_bounds__(256) k_gin_mlp(
    const float* __restrict__ Ain, float* __restrict__ Hout,
    const _Float16* __restrict__ w1t, const float* __restrict__ b1,
    const _Float16* __restrict__ w2t, const float* __restrict__ b2,
    float* __restrict__ stats, int ntiles) {
    __shared__ __align__(16) _Float16 sT[8][16 * 80];
    const int KP1 = (KIN < 32) ? 32 : KIN;  // padded K of first GEMM
    int wave = threadIdx.x >> 5;
    int lane = threadIdx.x & 31;
    int tile = blockIdx.x * 8 + wave;
    if (tile >= ntiles) return;
    int row_base = tile * 16;
    int n = lane & 15, hs = lane >> 4;
    _Float16* sp = sT[wave];

    // first linear: KIN -> 64, ReLU, stage to LDS as f16
    v16h a0 = load_a_g<KIN>(Ain, KIN, row_base, 0);
    v16h a1 = {};
    if (KIN > 32) a1 = load_a_g<KIN>(Ain, KIN, row_base, 32);
#pragma unroll
    for (int nt = 0; nt < 4; ++nt) {
        int n0 = nt * 16;
        v8f acc = {};
        v16h bf = load_b_t(w1t, KP1, 0, n0);
        acc = WMMA_F16(a0, bf, acc);
        if (KIN > 32) {
            v16h bf2 = load_b_t(w1t, KP1, 32, n0);
            acc = WMMA_F16(a1, bf2, acc);
        }
        float bias = b1[n0 + n];
#pragma unroll
        for (int v = 0; v < 8; ++v) {
            float t = acc[v] + bias;
            t = t > 0.0f ? t : 0.0f;
            sp[(v + 8 * hs) * 80 + n0 + n] = (_Float16)t;
        }
    }
    LDS_FENCE();

    // second linear: 64 -> 64, write pre-BN output, accumulate column sums/sumsq
    v16h c0 = load_a_s(sp, 0);
    v16h c1 = load_a_s(sp, 32);
    float* op = Hout + (size_t)row_base * 64;
#pragma unroll
    for (int nt = 0; nt < 4; ++nt) {
        int n0 = nt * 16;
        v8f acc = {};
        v16h bf = load_b_t(w2t, 64, 0, n0);
        acc = WMMA_F16(c0, bf, acc);
        v16h bf2 = load_b_t(w2t, 64, 32, n0);
        acc = WMMA_F16(c1, bf2, acc);
        float bias = b2[n0 + n];
        float cs = 0.0f, cq = 0.0f;
#pragma unroll
        for (int v = 0; v < 8; ++v) {
            float t = acc[v] + bias;
            op[(v + 8 * hs) * 64 + n0 + n] = t;
            cs += t;
            cq += t * t;
        }
        cs += __shfl_xor(cs, 16, 32);
        cq += __shfl_xor(cq, 16, 32);
        if (lane < 16) {
            atomicAdd(&stats[n0 + n], cs);
            atomicAdd(&stats[64 + n0 + n], cq);
        }
    }
}

// fold batch stats + gamma/beta into scale/shift
__global__ void k_bnfin(const float* __restrict__ stats, const float* __restrict__ g,
                        const float* __restrict__ be, float* __restrict__ coef, float invN) {
    int c = threadIdx.x;
    if (c >= 64) return;
    float mu = stats[c] * invN;
    float var = stats[64 + c] * invN - mu * mu;
    float sc = g[c] * rsqrtf(var + 1e-5f);
    coef[c] = sc;
    coef[64 + c] = be[c] - mu * sc;
}

__global__ void k_bnapply(const float* __restrict__ hpre, const float* __restrict__ coef,
                          float* __restrict__ h, long n) {
    long i = (long)blockIdx.x * blockDim.x + threadIdx.x;
    long stride = (long)gridDim.x * blockDim.x;
    for (; i < n; i += stride) {
        int c = (int)(i & 63);
        float t = hpre[i] * coef[c] + coef[64 + c];
        h[i] = fmaxf(t, 0.0f);
    }
}

// global mean pool, phase 1: atomic sums + counts
__global__ void k_pool(const float* __restrict__ h, const int* __restrict__ batch,
                       float* __restrict__ gsum, float* __restrict__ gcnt, int N_) {
    long t = (long)blockIdx.x * blockDim.x + threadIdx.x;
    int node = (int)(t >> 4);
    int q = (int)(t & 15);
    if (node >= N_) return;
    int b = batch[node];
    const float4 v = *(const float4*)(h + (size_t)node * 64 + q * 4);
    float* gp = gsum + b * 64 + q * 4;
    atomicAdd(gp + 0, v.x);
    atomicAdd(gp + 1, v.y);
    atomicAdd(gp + 2, v.z);
    atomicAdd(gp + 3, v.w);
    if (q == 0) atomicAdd(&gcnt[b], 1.0f);
}

__global__ void k_hgraph(const float* __restrict__ gsum, const float* __restrict__ gcnt,
                         float* __restrict__ hg) {
    int i = blockIdx.x * blockDim.x + threadIdx.x;
    if (i >= 64 * 64) return;
    int b = i >> 6;
    hg[i] = gsum[i] / fmaxf(gcnt[b], 1.0f);
}

// feat[b,c,:] = concat(hgraph[b], h[cand[b,c]])
__global__ void k_feat(const float* __restrict__ hg, const float* __restrict__ h,
                       const int* __restrict__ cand, float* __restrict__ feat) {
    int i = blockIdx.x * blockDim.x + threadIdx.x;
    if (i >= 64 * 32 * 128) return;
    int col = i & 127;
    int rc = i >> 7;
    feat[i] = (col < 64) ? hg[(rc >> 5) * 64 + col]
                         : h[(size_t)cand[rc] * 64 + (col - 64)];
}

// actor + critic heads: 2H -> 64 (tanh) -> 64 (tanh) -> 1, WMMA tiles of 16 rows
__global__ void __launch_bounds__(256) k_heads(
    const float* __restrict__ feat,
    const _Float16* aw0t, const float* ab0, const _Float16* aw1t, const float* ab1,
    const float* aw2, const float* ab2,
    const _Float16* cw0t, const float* cb0, const _Float16* cw1t, const float* cb1,
    const float* cw2, const float* cb2,
    float* __restrict__ logits, float* __restrict__ vvals, int nrows) {
    __shared__ __align__(16) _Float16 sbuf[8][2][16 * 80];
    int wave = threadIdx.x >> 5;
    int lane = threadIdx.x & 31;
    int tile = blockIdx.x * 8 + wave;
    if (tile * 16 >= nrows) return;
    int row_base = tile * 16;
    int n = lane & 15, hs = lane >> 4;

    v16h af[4];
#pragma unroll
    for (int kc = 0; kc < 4; ++kc) af[kc] = load_a_g<128>(feat, 128, row_base, kc * 32);

    for (int p = 0; p < 2; ++p) {
        const _Float16* w0 = p ? cw0t : aw0t;
        const float* bb0 = p ? cb0 : ab0;
        const _Float16* w1 = p ? cw1t : aw1t;
        const float* bb1 = p ? cb1 : ab1;
        const float* w2 = p ? cw2 : aw2;
        const float* bb2 = p ? cb2 : ab2;
        float* outp = p ? vvals : logits;
        _Float16* s0 = sbuf[wave][0];
        _Float16* s1 = sbuf[wave][1];

        // L0: 128 -> 64, tanh
#pragma unroll
        for (int nt = 0; nt < 4; ++nt) {
            int n0 = nt * 16;
            v8f acc = {};
#pragma unroll
            for (int kc = 0; kc < 4; ++kc) {
                v16h bf = load_b_t(w0, 128, kc * 32, n0);
                acc = WMMA_F16(af[kc], bf, acc);
            }
            float bias = bb0[n0 + n];
#pragma unroll
            for (int v = 0; v < 8; ++v)
                s0[(v + 8 * hs) * 80 + n0 + n] = (_Float16)fast_tanh(acc[v] + bias);
        }
        LDS_FENCE();

        // L1: 64 -> 64, tanh
        v16h c0 = load_a_s(s0, 0);
        v16h c1 = load_a_s(s0, 32);
#pragma unroll
        for (int nt = 0; nt < 4; ++nt) {
            int n0 = nt * 16;
            v8f acc = {};
            v16h bf = load_b_t(w1, 64, 0, n0);
            acc = WMMA_F16(c0, bf, acc);
            v16h bf2 = load_b_t(w1, 64, 32, n0);
            acc = WMMA_F16(c1, bf2, acc);
            float bias = bb1[n0 + n];
#pragma unroll
            for (int v = 0; v < 8; ++v)
                s1[(v + 8 * hs) * 80 + n0 + n] = (_Float16)fast_tanh(acc[v] + bias);
        }
        LDS_FENCE();

        // L2: 64 -> 1, lanes 0..15 each handle one row (vector LDS reads)
        if (lane < 16) {
            float sum = bb2[0];
            const v8h* rp = (const v8h*)(s1 + lane * 80);
#pragma unroll
            for (int c8 = 0; c8 < 8; ++c8) {
                v8h t = rp[c8];
#pragma unroll
                for (int j = 0; j < 8; ++j) sum += (float)t[j] * w2[c8 * 8 + j];
            }
            outp[row_base + lane] = sum;
        }
        LDS_FENCE();
    }
}

// per-graph log-softmax over C=32 candidates + append values
__global__ void k_out(const float* __restrict__ logits, const float* __restrict__ vvals,
                      float* __restrict__ out) {
    int b = blockIdx.x;
    int c = threadIdx.x;  // blockDim = 32
    float l = logits[b * 32 + c];
    float mx = l;
#pragma unroll
    for (int o = 16; o > 0; o >>= 1) mx = fmaxf(mx, __shfl_xor(mx, o, 32));
    float e = __expf(l - mx);
    float s = e;
#pragma unroll
    for (int o = 16; o > 0; o >>= 1) s += __shfl_xor(s, o, 32);
    out[b * 64 + c] = l - mx - __logf(s);
    out[b * 64 + 32 + c] = vvals[b * 32 + c];
}

extern "C" void kernel_launch(void* const* d_in, const int* in_sizes, int n_in,
                              void* d_out, int out_size, void* d_ws, size_t ws_size,
                              hipStream_t stream) {
    const float* x   = (const float*)d_in[0];
    const int* ei    = (const int*)d_in[1];
    const int* batch = (const int*)d_in[2];
    const int* cand  = (const int*)d_in[3];
    const float* w11 = (const float*)d_in[4];
    const float* b11 = (const float*)d_in[5];
    const float* w12 = (const float*)d_in[6];
    const float* b12 = (const float*)d_in[7];
    const float* w21 = (const float*)d_in[8];
    const float* b21 = (const float*)d_in[9];
    const float* w22 = (const float*)d_in[10];
    const float* b22 = (const float*)d_in[11];
    const float* g1  = (const float*)d_in[12];
    const float* be1 = (const float*)d_in[13];
    const float* g2  = (const float*)d_in[14];
    const float* be2 = (const float*)d_in[15];
    const float* aw0 = (const float*)d_in[16];
    const float* ab0 = (const float*)d_in[17];
    const float* aw1 = (const float*)d_in[18];
    const float* ab1 = (const float*)d_in[19];
    const float* aw2 = (const float*)d_in[20];
    const float* ab2 = (const float*)d_in[21];
    const float* cw0 = (const float*)d_in[22];
    const float* cb0 = (const float*)d_in[23];
    const float* cw1 = (const float*)d_in[24];
    const float* cb1 = (const float*)d_in[25];
    const float* cw2 = (const float*)d_in[26];
    const float* cb2 = (const float*)d_in[27];

    const int N = in_sizes[0] / 16;
    const int E = in_sizes[1] / 2;
    const int B = 64, C = 32;
    const int ntiles = (N + 15) / 16;

    // workspace carve-up (floats first, then 16B-aligned f16 weight region)
    float* ws     = (float*)d_ws;
    float* bufA   = ws;                               // N*64 : agg2 / pre-BN (in-place)
    float* bufB   = bufA + (size_t)N * 64;            // N*64 : post-BN h
    float* bufC   = bufB + (size_t)N * 64;            // N*16 : layer-1 agg
    float* stats  = bufC + (size_t)N * 16;            // 128
    float* coef   = stats + 128;                      // 128
    float* gsum   = coef + 128;                       // B*64
    float* gcnt   = gsum + B * 64;                    // B
    float* hg     = gcnt + B;                         // B*64
    float* feat   = hg + B * 64;                      // B*C*128
    float* logits = feat + (size_t)B * C * 128;       // B*C
    float* vvals  = logits + B * C;                   // B*C
    size_t foff   = (size_t)((vvals + B * C) - ws);
    foff          = (foff + 3) & ~(size_t)3;          // 16B align
    _Float16* w11t = (_Float16*)(ws + foff);          // 64*32
    _Float16* w12t = w11t + 64 * 32;                  // 64*64
    _Float16* w21t = w12t + 64 * 64;                  // 64*64
    _Float16* w22t = w21t + 64 * 64;                  // 64*64
    _Float16* aw0t = w22t + 64 * 64;                  // 64*128
    _Float16* aw1t = aw0t + 64 * 128;                 // 64*64
    _Float16* cw0t = aw1t + 64 * 64;                  // 64*128
    _Float16* cw1t = cw0t + 64 * 128;                 // 64*64
    float* out    = (float*)d_out;

    const int* esrc = ei;
    const int* edst = ei + E;
    const int TB = 256;

    // ---- weight prep (transpose + f16) ----
    k_wprep<<<(64 * 32 + TB - 1) / TB, TB, 0, stream>>>(w11, w11t, 16, 32, 64);
    k_wprep<<<(64 * 64 + TB - 1) / TB, TB, 0, stream>>>(w12, w12t, 64, 64, 64);
    k_wprep<<<(64 * 64 + TB - 1) / TB, TB, 0, stream>>>(w21, w21t, 64, 64, 64);
    k_wprep<<<(64 * 64 + TB - 1) / TB, TB, 0, stream>>>(w22, w22t, 64, 64, 64);
    k_wprep<<<(64 * 128 + TB - 1) / TB, TB, 0, stream>>>(aw0, aw0t, 128, 128, 64);
    k_wprep<<<(64 * 64 + TB - 1) / TB, TB, 0, stream>>>(aw1, aw1t, 64, 64, 64);
    k_wprep<<<(64 * 128 + TB - 1) / TB, TB, 0, stream>>>(cw0, cw0t, 128, 128, 64);
    k_wprep<<<(64 * 64 + TB - 1) / TB, TB, 0, stream>>>(cw1, cw1t, 64, 64, 64);

    // ---- layer 1 ----
    k_copy<<<4096, TB, 0, stream>>>(x, bufC, (long)N * 16);
    {
        long t = (long)E * 4;
        k_edge<<<(int)((t + TB - 1) / TB), TB, 0, stream>>>(x, esrc, edst, bufC, 16, 2, E);
    }
    k_zero<<<1, 128, 0, stream>>>(stats, 128);
    k_gin_mlp<16><<<(ntiles + 7) / 8, TB, 0, stream>>>(bufC, bufA, w11t, b11, w12t, b12, stats, ntiles);
    k_bnfin<<<1, 64, 0, stream>>>(stats, g1, be1, coef, 1.0f / (float)N);
    k_bnapply<<<4096, TB, 0, stream>>>(bufA, coef, bufB, (long)N * 64);

    // ---- layer 2 ----
    k_copy<<<4096, TB, 0, stream>>>(bufB, bufA, (long)N * 64);
    {
        long t = (long)E * 16;
        k_edge<<<(int)((t + TB - 1) / TB), TB, 0, stream>>>(bufB, esrc, edst, bufA, 64, 4, E);
    }
    k_zero<<<1, 128, 0, stream>>>(stats, 128);
    k_gin_mlp<64><<<(ntiles + 7) / 8, TB, 0, stream>>>(bufA, bufA, w21t, b21, w22t, b22, stats, ntiles);
    k_bnfin<<<1, 64, 0, stream>>>(stats, g2, be2, coef, 1.0f / (float)N);
    k_bnapply<<<4096, TB, 0, stream>>>(bufA, coef, bufB, (long)N * 64);

    // ---- pool + heads ----
    k_zero<<<(B * 64 + B + TB - 1) / TB, TB, 0, stream>>>(gsum, B * 64 + B);
    {
        long t = (long)N * 16;
        k_pool<<<(int)((t + TB - 1) / TB), TB, 0, stream>>>(bufB, batch, gsum, gcnt, N);
    }
    k_hgraph<<<(B * 64 + TB - 1) / TB, TB, 0, stream>>>(gsum, gcnt, hg);
    k_feat<<<(B * C * 128 + TB - 1) / TB, TB, 0, stream>>>(hg, bufB, cand, feat);
    {
        int htiles = (B * C + 15) / 16;  // 128
        k_heads<<<(htiles + 7) / 8, TB, 0, stream>>>(feat, aw0t, ab0, aw1t, ab1, aw2, ab2,
                                                     cw0t, cb0, cw1t, cb1, cw2, cb2,
                                                     logits, vvals, B * C);
    }
    k_out<<<B, 32, 0, stream>>>(logits, vvals, out);

    (void)n_in; (void)out_size; (void)ws_size;
}
